// Ansatz_fb_28174985462130
// MI455X (gfx1250) — matrix operand order repaired and checked by reading, not compiled, not run on hardware
//
#include <hip/hip_runtime.h>
#include <hip/hip_bf16.h>
#include <math.h>
#include <stdint.h>

// ---------------------------------------------------------------------------
// FermiNet-style ansatz forward pass for MI455X (gfx1250), wave32 + WMMA bf16.
// Heavy GEMMs: v_wmma_f32_16x16x32_bf16 with B staged through LDS via
// gfx1250 async loads (global_load_async_to_lds_b128 + s_wait_asynccnt).
// ---------------------------------------------------------------------------

typedef __attribute__((ext_vector_type(8)))  __bf16 v8bf;
typedef __attribute__((ext_vector_type(16))) __bf16 v16bf;
typedef __attribute__((ext_vector_type(8)))  float  v8f;

#define WMMA_BF16(A,B,C) \
  __builtin_amdgcn_wmma_f32_16x16x32_bf16(false,(A),false,(B),(short)0,(C),false,false)

constexpr int NB   = 4096;
constexpr int NE   = 14;
constexpr int NU   = 7;
constexpr int SV   = 256;
constexpr int PV   = 32;
constexpr int SVOUT= 832;   // 3*SV + 2*PV
constexpr int NDET = 16;
constexpr int NFB  = 3;
constexpr int NORB = NU*NDET; // 112
constexpr float AZ1 = 1.4f;

// ---- A-tile loader: 16x32 bf16 A tile for v_wmma_*_16x16x32, this file's K order (B matches) -------
// lane holds row M = lane%16; elements 0..7  = K in [half*16,  half*16+8)
//                             elements 8..15 = K in [half*16+8, half*16+16)
static __device__ inline v16bf load_a16(const __bf16* rowp, int half) {
  v8bf lo = *(const v8bf*)(rowp + half * 16);
  v8bf hi = *(const v8bf*)(rowp + half * 16 + 8);
  v16bf a;
#pragma unroll
  for (int i = 0; i < 8; ++i) { a[i] = lo[i]; a[8 + i] = hi[i]; }
  return a;
}

static __device__ inline v8f vzero8() {
  v8f z;
#pragma unroll
  for (int i = 0; i < 8; ++i) z[i] = 0.0f;
  return z;
}

// ---- CDNA5 async global->LDS copy (16B per lane, ASYNCcnt-tracked) ---------
static __device__ inline void async_copy16(unsigned lds_byte, const __bf16* gsrc) {
  unsigned long long ga = (unsigned long long)(uintptr_t)gsrc;
  asm volatile("global_load_async_to_lds_b128 %0, %1, off"
               :: "v"(lds_byte), "v"(ga) : "memory");
}
static __device__ inline void wait_async0() {
  asm volatile("s_wait_asynccnt 0x0" ::: "memory");
}

// ---------------------------------------------------------------------------
// Weight packing: fp32 [K][N] row-major  ->  bf16 [N][K] (B operand is a
// contiguous 16-element K-strip per lane when stored N-major).
// ---------------------------------------------------------------------------
__global__ void k_pack(const float* __restrict__ src, __bf16* __restrict__ dst,
                       int K, int N) {
  int t = blockIdx.x * 256 + threadIdx.x;
  if (t >= K * N) return;
  int k = t / N, n = t % N;
  dst[(size_t)n * K + k] = (__bf16)src[t];
}

// ---------------------------------------------------------------------------
// Geometry: pair features [dx,dy,dz,len] with the +eye trick on the diagonal.
// ---------------------------------------------------------------------------
__global__ void k_pin(const float* __restrict__ r, float* __restrict__ pin) {
  int t = blockIdx.x * 256 + threadIdx.x;
  if (t >= NB * NE * NE) return;
  int b = t / (NE * NE);
  int ij = t % (NE * NE);
  int i = ij / NE, j = ij % NE;
  const float* rj = r + ((size_t)b * NE + j) * 3;
  const float* ri = r + ((size_t)b * NE + i) * 3;
  float dx = rj[0] - ri[0], dy = rj[1] - ri[1], dz = rj[2] - ri[2];
  float d = (i == j) ? 1.0f : 0.0f;
  float len = sqrtf((dx + d) * (dx + d) + (dy + d) * (dy + d) + (dz + d) * (dz + d));
  float* o = pin + (size_t)t * 4;
  o[0] = dx; o[1] = dy; o[2] = dz; o[3] = len;
}

// single-electron features [ra0, |ra0|, ra1, |ra1|] and envelope sum_a exp(-|ra|)
__global__ void k_sin(const float* __restrict__ r, float* __restrict__ sin0,
                      float* __restrict__ env) {
  int t = blockIdx.x * 256 + threadIdx.x;
  if (t >= NB * NE) return;
  const float* rp = r + (size_t)t * 3;
  float x = rp[0], y = rp[1], z = rp[2];
  float l0 = sqrtf(x * x + y * y + z * z);
  float z1 = z - AZ1;
  float l1 = sqrtf(x * x + y * y + z1 * z1);
  float* o = sin0 + (size_t)t * 8;
  o[0] = x; o[1] = y; o[2] = z;  o[3] = l0;
  o[4] = x; o[5] = y; o[6] = z1; o[7] = l1;
  env[t] = expf(-l0) + expf(-l1);
}

__global__ void k_means_in_s(const float* __restrict__ sin0,
                             float* __restrict__ muin, float* __restrict__ mdin) {
  int t = blockIdx.x * 256 + threadIdx.x;
  if (t >= NB * 8) return;
  int b = t / 8, k = t % 8;
  float su = 0.f, sd = 0.f;
  for (int e = 0; e < NU; ++e)  su += sin0[((size_t)b * NE + e) * 8 + k];
  for (int e = NU; e < NE; ++e) sd += sin0[((size_t)b * NE + e) * 8 + k];
  muin[t] = su * (1.0f / NU);
  mdin[t] = sd * (1.0f / NU);
}

__global__ void k_means_in_p(const float* __restrict__ pin,
                             float* __restrict__ puin, float* __restrict__ pdin) {
  int t = blockIdx.x * 256 + threadIdx.x;
  if (t >= NB * NE * 4) return;
  int b = t / (NE * 4);
  int jk = t % (NE * 4);
  int j = jk / 4, k = jk % 4;
  float su = 0.f, sd = 0.f;
  for (int i = 0; i < NU; ++i)  su += pin[(((size_t)b * NE + i) * NE + j) * 4 + k];
  for (int i = NU; i < NE; ++i) sd += pin[(((size_t)b * NE + i) * NE + j) * 4 + k];
  puin[t] = su * (1.0f / NU);
  pdin[t] = sd * (1.0f / NU);
}

// initial s layer: (nb*14, 32) @ (32, 256) -> tanh -> bf16  (small; plain VALU)
__global__ void k_s0(const float* __restrict__ sin0, const float* __restrict__ muin,
                     const float* __restrict__ mdin, const float* __restrict__ puin,
                     const float* __restrict__ pdin, const float* __restrict__ w0,
                     const float* __restrict__ b0, __bf16* __restrict__ sv) {
  int t = blockIdx.x * 256 + threadIdx.x;
  if (t >= NB * NE * SV) return;
  int n = t % SV;
  int be = t / SV;
  int b = be / NE;
  float blk[32];
  const float* s  = sin0 + (size_t)be * 8;
  const float* m1 = muin + (size_t)b * 8;
  const float* m2 = mdin + (size_t)b * 8;
  const float* q1 = puin + (size_t)be * 4;
  const float* q2 = pdin + (size_t)be * 4;
#pragma unroll
  for (int k = 0; k < 8; ++k) { blk[k] = s[k]; blk[8 + k] = m1[k]; blk[16 + k] = m2[k]; }
#pragma unroll
  for (int k = 0; k < 4; ++k) { blk[24 + k] = q1[k]; blk[28 + k] = q2[k]; }
  float acc = b0[n];
#pragma unroll
  for (int k = 0; k < 32; ++k) acc += blk[k] * w0[(size_t)k * SV + n];
  sv[t] = (__bf16)tanhf(acc);
}

// initial p layer: (nb*196, 4) @ (4, 32) -> tanh -> bf16
__global__ void k_p0(const float* __restrict__ pin, const float* __restrict__ w0,
                     const float* __restrict__ b0, __bf16* __restrict__ pv) {
  int t = blockIdx.x * 256 + threadIdx.x;
  if (t >= NB * NE * NE * PV) return;
  int n = t % PV;
  int bij = t / PV;
  const float* p = pin + (size_t)bij * 4;
  float acc = b0[n];
#pragma unroll
  for (int k = 0; k < 4; ++k) acc += p[k] * w0[(size_t)k * PV + n];
  pv[t] = (__bf16)tanhf(acc);
}

// per-layer spin means of s stream (bf16 in/out, fp32 accumulate)
__global__ void k_means_s(const __bf16* __restrict__ sv, __bf16* __restrict__ mu,
                          __bf16* __restrict__ md) {
  int t = blockIdx.x * 256 + threadIdx.x;
  if (t >= NB * SV) return;
  int b = t / SV, k = t % SV;
  float su = 0.f, sd = 0.f;
  for (int e = 0; e < NU; ++e)  su += (float)sv[((size_t)b * NE + e) * SV + k];
  for (int e = NU; e < NE; ++e) sd += (float)sv[((size_t)b * NE + e) * SV + k];
  mu[t] = (__bf16)(su * (1.0f / NU));
  md[t] = (__bf16)(sd * (1.0f / NU));
}

// per-layer spin means of p stream (mean over first electron index)
__global__ void k_means_p(const __bf16* __restrict__ pv, __bf16* __restrict__ pu,
                          __bf16* __restrict__ pd) {
  int t = blockIdx.x * 256 + threadIdx.x;
  if (t >= NB * NE * PV) return;
  int b = t / (NE * PV);
  int jk = t % (NE * PV);
  int j = jk / PV, k = jk % PV;
  float su = 0.f, sd = 0.f;
  for (int i = 0; i < NU; ++i)  su += (float)pv[(((size_t)b * NE + i) * NE + j) * PV + k];
  for (int i = NU; i < NE; ++i) sd += (float)pv[(((size_t)b * NE + i) * NE + j) * PV + k];
  pu[t] = (__bf16)(su * (1.0f / NU));
  pd[t] = (__bf16)(sd * (1.0f / NU));
}

// ---------------------------------------------------------------------------
// Main s-stream GEMM: (57344 x 832) @ (832 x 256).
// Wave = 16 rows x 256 cols (16 WMMA acc tiles); workgroup = 8 waves = 128 rows.
// B chunk (256 cols x 32 K = 16KB bf16) double-buffered in LDS, filled with
// async global->LDS loads while the previous chunk computes. grid = (448).
// ---------------------------------------------------------------------------
__global__ __launch_bounds__(256) void k_sgemm(
    const __bf16* __restrict__ svc, const __bf16* __restrict__ mu,
    const __bf16* __restrict__ md, const __bf16* __restrict__ pu,
    const __bf16* __restrict__ pd, const __bf16* __restrict__ wT,
    const float* __restrict__ bias, __bf16* __restrict__ svn, int residual) {
  __shared__ v16bf lbufv[2][512];   // 2 x 16KB: [buf][n*2 + khalf]
  const int lane = threadIdx.x & 31, wave = threadIdx.x >> 5;
  const int half = lane >> 4, l16 = lane & 15;
  const int m0 = (blockIdx.x * 8 + wave) * 16;
  const int lrow = m0 + l16;
  const int lb = lrow / NE;
  const __bf16* pa_sv = svc + (size_t)lrow * SV;
  const __bf16* pa_mu = mu  + (size_t)lb  * SV;
  const __bf16* pa_md = md  + (size_t)lb  * SV;
  const __bf16* pa_pu = pu  + (size_t)lrow * PV;
  const __bf16* pa_pd = pd  + (size_t)lrow * PV;
  const unsigned lbase = (unsigned)(uintptr_t)(&lbufv[0][0]);

  // this thread stages 4 x 16B per chunk: rows n = wave*32 + (lane/4)*? laid out
  // as 32 pieces of 512B; piece p = wave*4+i -> rows p*8..p*8+7, lane covers 16B.
  auto stage = [&](int cc) {
#pragma unroll
    for (int i = 0; i < 4; ++i) {
      int p = wave * 4 + i;
      int n = p * 8 + (lane >> 2);
      int q = lane & 3;
      unsigned dst = lbase + (unsigned)((cc & 1) * 16384 + n * 64 + q * 16);
      async_copy16(dst, wT + (size_t)n * SVOUT + cc * 32 + q * 8);
    }
  };

  v8f acc[16];
#pragma unroll
  for (int t = 0; t < 16; ++t) acc[t] = vzero8();

  stage(0);
  for (int c = 0; c < 26; ++c) {
    wait_async0();       // my stage of chunk c landed in LDS
    __syncthreads();     // everyone's stage landed; prev-chunk readers done
    if (c + 1 < 26) stage(c + 1);
    const __bf16* pa;
    if (c < 8)        pa = pa_sv + c * 32;
    else if (c < 16)  pa = pa_mu + (c - 8) * 32;
    else if (c < 24)  pa = pa_md + (c - 16) * 32;
    else if (c == 24) pa = pa_pu;
    else              pa = pa_pd;
    v16bf a = load_a16(pa, half);
#pragma unroll
    for (int t = 0; t < 16; ++t) {
      v16bf bm = lbufv[c & 1][(t * 16 + l16) * 2 + half];
      acc[t] = WMMA_BF16(a, bm, acc[t]);
    }
  }

#pragma unroll
  for (int t = 0; t < 16; ++t) {
    int col = t * 16 + l16;
    float bv = bias[col];
#pragma unroll
    for (int v = 0; v < 8; ++v) {
      int row = m0 + v + half * 8;
      float val = tanhf(acc[t][v] + bv);
      if (residual) val += (float)svc[(size_t)row * SV + col];
      svn[(size_t)row * SV + col] = (__bf16)val;
    }
  }
}

// ---------------------------------------------------------------------------
// p-stream GEMM (in place): (802816 x 32) @ (32 x 32), tanh + residual.
// Wave = 16 rows x full 32 cols -> safe in-place update.
// ---------------------------------------------------------------------------
__global__ __launch_bounds__(256) void k_pgemm(__bf16* __restrict__ pv,
                                               const __bf16* __restrict__ wT,
                                               const float* __restrict__ bias) {
  const int lane = threadIdx.x & 31, wave = threadIdx.x >> 5;
  const int half = lane >> 4, l16 = lane & 15;
  const int m0 = (blockIdx.x * 8 + wave) * 16;
  const int lrow = m0 + l16;
  v16bf a = load_a16(pv + (size_t)lrow * PV, half);
  v8f acc[2];
  acc[0] = vzero8(); acc[1] = vzero8();
#pragma unroll
  for (int t = 0; t < 2; ++t) {
    v16bf bm = *(const v16bf*)(wT + (size_t)(t * 16 + l16) * PV + half * 16);
    acc[t] = WMMA_BF16(a, bm, acc[t]);
  }
#pragma unroll
  for (int t = 0; t < 2; ++t) {
    int col = t * 16 + l16;
    float bv = bias[col];
#pragma unroll
    for (int v = 0; v < 8; ++v) {
      int row = m0 + v + half * 8;
      float res = (float)pv[(size_t)row * PV + col];
      pv[(size_t)row * PV + col] = (__bf16)(tanhf(acc[t][v] + bv) + res);
    }
  }
}

// ---------------------------------------------------------------------------
// Orbital GEMM: per spin (28672 x 256) @ (256 x 112), + bias, * envelope.
// Entire packed weight (112x256 bf16 = 56KB) async-staged into LDS once per
// workgroup; all 8 K-chunks then run out of LDS. grid = (224, 2).
// ---------------------------------------------------------------------------
__global__ __launch_bounds__(256) void k_orb(
    const __bf16* __restrict__ svf, const __bf16* __restrict__ wuT,
    const __bf16* __restrict__ wdT, const float* __restrict__ bu,
    const float* __restrict__ bd, const float* __restrict__ env,
    float* __restrict__ swu, float* __restrict__ swd) {
  __shared__ v16bf obuf[1792];   // 112*256 bf16 = 57344 B: [n*16 + c*2 + half]
  const int lane = threadIdx.x & 31, wave = threadIdx.x >> 5;
  const int half = lane >> 4, l16 = lane & 15;
  const int spin = blockIdx.y;
  const __bf16* wT   = spin ? wdT : wuT;
  const float*  bias = spin ? bd  : bu;
  float*        out  = spin ? swd : swu;

  const unsigned obase = (unsigned)(uintptr_t)(&obuf[0]);
#pragma unroll
  for (int i = 0; i < 14; ++i) {         // 112 x 512B pieces over 8 waves
    int j = wave * 14 + i;
    unsigned dst = obase + (unsigned)(j * 512 + lane * 16);
    async_copy16(dst, wT + (size_t)j * 256 + lane * 8);
  }
  wait_async0();
  __syncthreads();

  const int m0 = (blockIdx.x * 8 + wave) * 16;
  const int lrow = m0 + l16;
  const int lb = lrow / NU, le = lrow % NU;
  const __bf16* pa = svf + ((size_t)lb * NE + spin * NU + le) * SV;

  v8f acc[7];
#pragma unroll
  for (int t = 0; t < 7; ++t) acc[t] = vzero8();

  for (int c = 0; c < 8; ++c) {
    v16bf a = load_a16(pa + c * 32, half);
#pragma unroll
    for (int t = 0; t < 7; ++t) {
      v16bf bm = obuf[(t * 16 + l16) * 16 + c * 2 + half];
      acc[t] = WMMA_BF16(a, bm, acc[t]);
    }
  }

#pragma unroll
  for (int t = 0; t < 7; ++t) {
    int col = t * 16 + l16;
    float bv = bias[col];
#pragma unroll
    for (int v = 0; v < 8; ++v) {
      int row = m0 + v + half * 8;
      int rb = row / NU, re = row % NU;
      float ev = env[(size_t)rb * NE + spin * NU + re];
      out[(size_t)row * NORB + col] = (acc[t][v] + bv) * ev;
    }
  }
}

// ---------------------------------------------------------------------------
// slogdet of 7x7 (u and d combined) per (sample, det): LU w/ partial pivoting.
// M[o][e] = sw[b][e][o*16+d]  (transpose doesn't affect det; matches ref).
// ---------------------------------------------------------------------------
__global__ void k_det(const float* __restrict__ swu, const float* __restrict__ swd,
                      float* __restrict__ sgn, float* __restrict__ ld) {
  int t = blockIdx.x * 256 + threadIdx.x;
  if (t >= NB * NDET) return;
  int b = t / NDET, d = t % NDET;
  float sign = 1.0f, logd = 0.0f;
  for (int spin = 0; spin < 2; ++spin) {
    const float* sw = spin ? swd : swu;
    float M[NU][NU];
    for (int o = 0; o < NU; ++o)
      for (int e = 0; e < NU; ++e)
        M[o][e] = sw[((size_t)b * NU + e) * NORB + o * NDET + d];
    for (int k = 0; k < NU; ++k) {
      int p = k;
      float best = fabsf(M[k][k]);
      for (int i = k + 1; i < NU; ++i) {
        float v = fabsf(M[i][k]);
        if (v > best) { best = v; p = i; }
      }
      if (p != k) {
        for (int j = 0; j < NU; ++j) {
          float tmp = M[k][j]; M[k][j] = M[p][j]; M[p][j] = tmp;
        }
        sign = -sign;
      }
      float piv = M[k][k];
      if (piv < 0.0f) sign = -sign;
      logd += logf(fabsf(piv));
      float inv = 1.0f / piv;
      for (int i = k + 1; i < NU; ++i) {
        float f = M[i][k] * inv;
        for (int j = k + 1; j < NU; ++j) M[i][j] -= f * M[k][j];
      }
    }
  }
  sgn[t] = sign;
  ld[t] = logd;
}

// log-sum-exp over dets + weighted combine -> log|psi|
__global__ void k_final(const float* __restrict__ sgn, const float* __restrict__ ld,
                        const float* __restrict__ wf, float* __restrict__ out) {
  int b = blockIdx.x * 256 + threadIdx.x;
  if (b >= NB) return;
  float mx = -INFINITY;
  for (int d = 0; d < NDET; ++d) mx = fmaxf(mx, ld[b * NDET + d]);
  float s = 0.0f;
  for (int d = 0; d < NDET; ++d)
    s += sgn[b * NDET + d] * expf(ld[b * NDET + d] - mx) * wf[d];
  out[b] = logf(fabsf(s)) + mx;
}

// ---------------------------------------------------------------------------
// Workspace layout (bytes; every block size is a multiple of 256)
// ---------------------------------------------------------------------------
constexpr size_t SZ_SV  = (size_t)NB * NE * SV * 2;
constexpr size_t SZ_PV  = (size_t)NB * NE * NE * PV * 2;
constexpr size_t O_SVA = 0;
constexpr size_t O_SVB = O_SVA + SZ_SV;
constexpr size_t O_PV  = O_SVB + SZ_SV;
constexpr size_t O_MU  = O_PV  + SZ_PV;
constexpr size_t O_MD  = O_MU  + (size_t)NB * SV * 2;
constexpr size_t O_PU  = O_MD  + (size_t)NB * SV * 2;
constexpr size_t O_PD  = O_PU  + (size_t)NB * NE * PV * 2;
constexpr size_t O_SIN = O_PD  + (size_t)NB * NE * PV * 2;
constexpr size_t O_PIN = O_SIN + (size_t)NB * NE * 8 * 4;
constexpr size_t O_MUI = O_PIN + (size_t)NB * NE * NE * 4 * 4;
constexpr size_t O_MDI = O_MUI + (size_t)NB * 8 * 4;
constexpr size_t O_PUI = O_MDI + (size_t)NB * 8 * 4;
constexpr size_t O_PDI = O_PUI + (size_t)NB * NE * 4 * 4;
constexpr size_t O_ENV = O_PDI + (size_t)NB * NE * 4 * 4;
constexpr size_t O_SWU = O_ENV + (size_t)NB * NE * 4;
constexpr size_t O_SWD = O_SWU + (size_t)NB * NU * NORB * 4;
constexpr size_t O_SGN = O_SWD + (size_t)NB * NU * NORB * 4;
constexpr size_t O_LD  = O_SGN + (size_t)NB * NDET * 4;
constexpr size_t O_SWT = O_LD  + (size_t)NB * NDET * 4;
constexpr size_t O_VAT = O_SWT + (size_t)NFB * SV * SVOUT * 2;
constexpr size_t O_PWT = O_VAT + (size_t)SV * SVOUT * 2;
constexpr size_t O_WUT = O_PWT + (size_t)NFB * PV * PV * 2;
constexpr size_t O_WDT = O_WUT + (size_t)NORB * SV * 2;

extern "C" void kernel_launch(void* const* d_in, const int* in_sizes, int n_in,
                              void* d_out, int out_size, void* d_ws, size_t ws_size,
                              hipStream_t stream) {
  (void)in_sizes; (void)n_in; (void)out_size; (void)ws_size;
  const float* r    = (const float*)d_in[0];
  const float* s_w0 = (const float*)d_in[1];
  const float* s_b0 = (const float*)d_in[2];
  const float* s_w  = (const float*)d_in[3];
  const float* s_b  = (const float*)d_in[4];
  const float* p_w0 = (const float*)d_in[5];
  const float* p_b0 = (const float*)d_in[6];
  const float* p_w  = (const float*)d_in[7];
  const float* p_b  = (const float*)d_in[8];
  const float* va_w = (const float*)d_in[9];
  const float* va_b = (const float*)d_in[10];
  const float* wu_w = (const float*)d_in[11];
  const float* wu_b = (const float*)d_in[12];
  const float* wd_w = (const float*)d_in[13];
  const float* wd_b = (const float*)d_in[14];
  const float* wf_w = (const float*)d_in[15];

  char* ws = (char*)d_ws;
  __bf16* svA = (__bf16*)(ws + O_SVA);
  __bf16* svB = (__bf16*)(ws + O_SVB);
  __bf16* pv  = (__bf16*)(ws + O_PV);
  __bf16* mu  = (__bf16*)(ws + O_MU);
  __bf16* md  = (__bf16*)(ws + O_MD);
  __bf16* pu  = (__bf16*)(ws + O_PU);
  __bf16* pd  = (__bf16*)(ws + O_PD);
  float* sin0 = (float*)(ws + O_SIN);
  float* pin  = (float*)(ws + O_PIN);
  float* muin = (float*)(ws + O_MUI);
  float* mdin = (float*)(ws + O_MDI);
  float* puin = (float*)(ws + O_PUI);
  float* pdin = (float*)(ws + O_PDI);
  float* env  = (float*)(ws + O_ENV);
  float* swu  = (float*)(ws + O_SWU);
  float* swd  = (float*)(ws + O_SWD);
  float* sgn  = (float*)(ws + O_SGN);
  float* ld   = (float*)(ws + O_LD);
  __bf16* swT = (__bf16*)(ws + O_SWT);
  __bf16* vaT = (__bf16*)(ws + O_VAT);
  __bf16* pwT = (__bf16*)(ws + O_PWT);
  __bf16* wuT = (__bf16*)(ws + O_WUT);
  __bf16* wdT = (__bf16*)(ws + O_WDT);

  // ---- pack weights (fp32 KxN -> bf16 NxK) --------------------------------
  for (int i = 0; i < NFB; ++i)
    k_pack<<<(SVOUT * SV + 255) / 256, 256, 0, stream>>>(
        s_w + (size_t)i * SVOUT * SV, swT + (size_t)i * SV * SVOUT, SVOUT, SV);
  k_pack<<<(SVOUT * SV + 255) / 256, 256, 0, stream>>>(va_w, vaT, SVOUT, SV);
  for (int i = 0; i < NFB; ++i)
    k_pack<<<(PV * PV + 255) / 256, 256, 0, stream>>>(
        p_w + (size_t)i * PV * PV, pwT + (size_t)i * PV * PV, PV, PV);
  k_pack<<<(SV * NORB + 255) / 256, 256, 0, stream>>>(wu_w, wuT, SV, NORB);
  k_pack<<<(SV * NORB + 255) / 256, 256, 0, stream>>>(wd_w, wdT, SV, NORB);

  // ---- geometry + initial embeddings --------------------------------------
  k_pin<<<(NB * NE * NE + 255) / 256, 256, 0, stream>>>(r, pin);
  k_sin<<<(NB * NE + 255) / 256, 256, 0, stream>>>(r, sin0, env);
  k_means_in_s<<<(NB * 8 + 255) / 256, 256, 0, stream>>>(sin0, muin, mdin);
  k_means_in_p<<<(NB * NE * 4 + 255) / 256, 256, 0, stream>>>(pin, puin, pdin);
  k_s0<<<(NB * NE * SV + 255) / 256, 256, 0, stream>>>(sin0, muin, mdin, puin,
                                                       pdin, s_w0, s_b0, svA);
  k_p0<<<(NB * NE * NE * PV + 255) / 256, 256, 0, stream>>>(pin, p_w0, p_b0, pv);

  // ---- FermiBlock layers ---------------------------------------------------
  __bf16* cur = svA;
  __bf16* nxt = svB;
  for (int i = 0; i < NFB; ++i) {
    k_means_s<<<(NB * SV + 255) / 256, 256, 0, stream>>>(cur, mu, md);
    k_means_p<<<(NB * NE * PV + 255) / 256, 256, 0, stream>>>(pv, pu, pd);
    k_sgemm<<<NB * NE / 128, 256, 0, stream>>>(
        cur, mu, md, pu, pd, swT + (size_t)i * SV * SVOUT, s_b + (size_t)i * SV,
        nxt, 1);
    k_pgemm<<<NB * NE * NE / 128, 256, 0, stream>>>(
        pv, pwT + (size_t)i * PV * PV, p_b + (size_t)i * PV);
    __bf16* tmp = cur; cur = nxt; nxt = tmp;
  }

  // ---- final embedding (va, no residual) ----------------------------------
  k_means_s<<<(NB * SV + 255) / 256, 256, 0, stream>>>(cur, mu, md);
  k_means_p<<<(NB * NE * PV + 255) / 256, 256, 0, stream>>>(pv, pu, pd);
  k_sgemm<<<NB * NE / 128, 256, 0, stream>>>(cur, mu, md, pu, pd, vaT, va_b,
                                             nxt, 0);

  // ---- orbitals, determinants, combine ------------------------------------
  k_orb<<<dim3(NB * NU / 128, 2), 256, 0, stream>>>(nxt, wuT, wdT, wu_b, wd_b,
                                                    env, swu, swd);
  k_det<<<(NB * NDET + 255) / 256, 256, 0, stream>>>(swu, swd, sgn, ld);
  k_final<<<(NB + 255) / 256, 256, 0, stream>>>(sgn, ld, wf_w, (float*)d_out);
}
